// Mask2FormerLoss_34299608826006
// MI455X (gfx1250) — matrix operand order repaired and checked by reading, not compiled, not run on hardware
//
#include <hip/hip_runtime.h>
#include <hip/hip_bf16.h>

// ---------------------------------------------------------------------------
// Mask2Former loss, MI455X (gfx1250, wave32).
//   - cost einsum (dice inter + cubic-in-t focal decomposition) ->
//     v_wmma_f32_16x16x32_f16, unroll-by-2 ping-pong pipeline (no register
//     rotation moves, no in-loop guards -> no EXEC churn), K split across
//     4 waves with an LDS combine
//   - Hungarian LSA: one 128-thread block per (l,b), parallel min/argmin scans
//   - full-res matched-pair focal/dice: fused bilinear 128->512 upsample,
//     block reductions, global_prefetch on the streaming gt read
// ---------------------------------------------------------------------------

#define LN   3
#define BN   2
#define QN   100
#define QP   112     // Q padded to 7 WMMA tiles
#define NGT  20
#define NP   32      // N padded to 2 WMMA tiles
#define PDS  4096    // 64*64 downsample pixels
#define CC   2
#define HREZ 512
#define HSML 128
#define ALPHA_F 0.25f
#define KWAVES 4     // waves per cost block, each owns PDS/KWAVES of K
#define KSTEPS ((PDS/KWAVES)/32)   // 32 k-steps of 32 per wave

typedef _Float16 v16h __attribute__((ext_vector_type(16)));
typedef _Float16 v8h  __attribute__((ext_vector_type(8)));
typedef float    v8f  __attribute__((ext_vector_type(8)));

// ---- workspace byte offsets (16B aligned) ----
#define OFF_T1    (size_t)0                         // [B][NP][P] f16
#define OFF_T2    (OFF_T1 + (size_t)BN*NP*PDS*2)
#define OFF_T3    (OFF_T2 + (size_t)BN*NP*PDS*2)
#define OFF_TSUM  (OFF_T3 + (size_t)BN*NP*PDS*2)    // [B][NP] f32
#define OFF_PROB  (OFF_TSUM + (size_t)BN*NP*4 + 128)
#define OFF_K1    (OFF_PROB + (size_t)BN*QP*PDS*2)
#define OFF_K2    (OFF_K1  + (size_t)BN*QP*PDS*2)
#define OFF_K3    (OFF_K2  + (size_t)BN*QP*PDS*2)
#define OFF_QSC   (OFF_K3  + (size_t)BN*QP*PDS*2)   // [B][QP][2] f32 (psum, c0)
#define OFF_COST  (OFF_QSC + (size_t)BN*QP*2*4 + 128) // [L][B][Q][N] f32
#define OFF_MATCH (OFF_COST + (size_t)LN*BN*QN*NGT*4)  // [L][B][N] int (gt n -> query q)
#define OFF_PAIR  (OFF_MATCH + (size_t)LN*BN*NGT*4)    // [L][B][N][4] f32

// ===========================================================================
// 1) gt masks: 512x512 int -> 64x64 f32 box-avg; emit t, t^2, t^3 (f16, padded
//    to NP rows) and per-mask pixel sum.
// ===========================================================================
__global__ __launch_bounds__(256)
void prep_gt_kernel(const int* __restrict__ gt, _Float16* __restrict__ t1,
                    _Float16* __restrict__ t2, _Float16* __restrict__ t3,
                    float* __restrict__ tsum)
{
    const int blk = blockIdx.x;
    const int b = blk / NP, n = blk % NP;
    const int tid = threadIdx.x;
    const size_t ob = (size_t)(b*NP + n) * PDS;
    const int* src = gt + ((size_t)b*NGT + n) * (HREZ*HREZ);
    float acc = 0.f;
    for (int it = 0; it < PDS/256; ++it) {
        const int p = tid + it*256;
        float t = 0.f;
        if (n < NGT) {
            const int oy = p >> 6, ox = p & 63;
            const int* r = src + (size_t)oy*8*HREZ + ox*8;
            __builtin_prefetch(r + 16*HREZ, 0, 0);   // next block row
            int s = 0;
            #pragma unroll
            for (int dy = 0; dy < 8; ++dy)
                #pragma unroll
                for (int dx = 0; dx < 8; ++dx) s += r[dy*HREZ + dx];
            t = (float)s * (1.0f/64.0f);
        }
        t1[ob+p] = (_Float16)t;
        t2[ob+p] = (_Float16)(t*t);
        t3[ob+p] = (_Float16)(t*t*t);
        acc += t;
    }
    __shared__ float sr[256];
    sr[tid] = acc; __syncthreads();
    for (int s = 128; s > 0; s >>= 1) {
        if (tid < s) sr[tid] += sr[tid+s];
        __syncthreads();
    }
    if (tid == 0) tsum[b*NP + n] = sr[0];
}

// ===========================================================================
// 2) pred masks (one layer): 128x128 -> 64x64 (2x2 avg; approximates the
//    128->512->64 bilinear chain), then per-pixel cubic focal coefficients:
//    focal(x,t) = a*(pr - s t)^2 (A - x t) = k0 + k1 t + k2 t^2 + k3 t^3
//    with pr=sigmoid(x), s=2pr-1, A=max(x,0)+log1p(exp(-|x|)).
//    Emits prob,k1,k2,k3 (f16, Q padded to QP) + per-query (prob_sum, sum k0).
// ===========================================================================
__global__ __launch_bounds__(256)
void prep_pred_kernel(int l, const float* __restrict__ pm,
                      _Float16* __restrict__ prob, _Float16* __restrict__ k1,
                      _Float16* __restrict__ k2,  _Float16* __restrict__ k3,
                      float* __restrict__ qscal)
{
    const int blk = blockIdx.x;
    const int b = blk / QP, q = blk % QP;
    const int tid = threadIdx.x;
    const float* src = pm + (((size_t)l*BN + b)*QN + q) * (HSML*HSML);
    const size_t ob = (size_t)(b*QP + q) * PDS;
    float psum = 0.f, c0 = 0.f;
    for (int it = 0; it < PDS/256; ++it) {
        const int p = tid + it*256;
        _Float16 hpr = (_Float16)0.f, h1 = (_Float16)0.f,
                 h2  = (_Float16)0.f, h3 = (_Float16)0.f;
        if (q < QN) {
            const int oy = p >> 6, ox = p & 63;
            const float* r0 = src + (size_t)(2*oy)*HSML + 2*ox;
            const float x = 0.25f * (r0[0] + r0[1] + r0[HSML] + r0[HSML+1]);
            const float pr = 1.f / (1.f + __expf(-x));
            const float A  = fmaxf(x, 0.f) + log1pf(__expf(-fabsf(x)));
            const float s  = 2.f*pr - 1.f;
            hpr = (_Float16)pr;
            h1  = (_Float16)(ALPHA_F * (-2.f*pr*s*A - pr*pr*x));
            h2  = (_Float16)(ALPHA_F * ( s*s*A + 2.f*pr*s*x));
            h3  = (_Float16)(ALPHA_F * (-s*s*x));
            psum += pr;
            c0   += ALPHA_F * pr*pr*A;
        }
        prob[ob+p] = hpr; k1[ob+p] = h1; k2[ob+p] = h2; k3[ob+p] = h3;
    }
    __shared__ float sr[512];
    sr[tid*2+0] = psum; sr[tid*2+1] = c0; __syncthreads();
    for (int s = 128; s > 0; s >>= 1) {
        if (tid < s) { sr[tid*2+0] += sr[(tid+s)*2+0]; sr[tid*2+1] += sr[(tid+s)*2+1]; }
        __syncthreads();
    }
    if (tid == 0) { qscal[(b*QP+q)*2+0] = sr[0]; qscal[(b*QP+q)*2+1] = sr[1]; }
}

// ===========================================================================
// 3) Cost matrix via WMMA.  4 accumulations per 16x16 tile over K=4096:
//    accP = prob @ t, acc1 = k1 @ t, acc2 = k2 @ t^2, acc3 = k3 @ t^3.
//    K split across KWAVES waves; each wave runs an unroll-by-2 ping-pong
//    pipeline with constant trip count: loads *define* each tile-set's
//    registers (no rotation moves) and there are no in-loop guards
//    (no EXEC manipulation around the WMMAs).
//    A-layout per ISA: lane M = lane&15, k-halves split by lane>=16.
//    B-layout: lane N = lane&15, same K order as A (load_a16) (t stored [n][p]).
// ===========================================================================
__device__ __forceinline__ v16h load_a16(const _Float16* base, int k0, int koffA)
{
    v8h lo = *(const v8h*)(base + k0 + koffA);
    v8h hi = *(const v8h*)(base + k0 + 16 + koffA);
    return __builtin_shufflevector(lo, hi, 0,1,2,3,4,5,6,7,8,9,10,11,12,13,14,15);
}

struct TileSet { v16h aP, a1, a2, a3, b1, b2, b3; };

__device__ __forceinline__ TileSet load_tiles(
    const _Float16* Ap, const _Float16* A1, const _Float16* A2, const _Float16* A3,
    const _Float16* B1, const _Float16* B2, const _Float16* B3,
    int k0, int koffA, int koffB)
{
    TileSet s;
    s.aP = load_a16(Ap, k0, koffA);
    s.a1 = load_a16(A1, k0, koffA);
    s.a2 = load_a16(A2, k0, koffA);
    s.a3 = load_a16(A3, k0, koffA);
    s.b1 = load_a16(B1, k0, koffA);
    s.b2 = load_a16(B2, k0, koffA);
    s.b3 = load_a16(B3, k0, koffA);
    return s;
}

__device__ __forceinline__ void wmma4(const TileSet& s, v8f& accP, v8f& acc1,
                                      v8f& acc2, v8f& acc3)
{
    accP = __builtin_amdgcn_wmma_f32_16x16x32_f16(false, s.aP, false, s.b1, (short)0, accP, false, false);
    acc1 = __builtin_amdgcn_wmma_f32_16x16x32_f16(false, s.a1, false, s.b1, (short)0, acc1, false, false);
    acc2 = __builtin_amdgcn_wmma_f32_16x16x32_f16(false, s.a2, false, s.b2, (short)0, acc2, false, false);
    acc3 = __builtin_amdgcn_wmma_f32_16x16x32_f16(false, s.a3, false, s.b3, (short)0, acc3, false, false);
}

__global__ __launch_bounds__(32*KWAVES)
void cost_wmma_kernel(int l,
                      const _Float16* __restrict__ prob, const _Float16* __restrict__ k1,
                      const _Float16* __restrict__ k2,  const _Float16* __restrict__ k3,
                      const _Float16* __restrict__ t1,  const _Float16* __restrict__ t2,
                      const _Float16* __restrict__ t3,
                      const float* __restrict__ qscal, const float* __restrict__ tsum,
                      const float* __restrict__ logits, const int* __restrict__ labels,
                      float* __restrict__ cost)
{
    const int tid  = threadIdx.x;
    const int wave = tid >> 5;            // 0..KWAVES-1: K-slice owner
    const int lane = tid & 31;
    const int tm = blockIdx.x;            // 0..6  (query tiles)
    const int tn = blockIdx.y;            // 0..1  (gt tiles)
    const int b  = blockIdx.z;

    const int mrow   = tm*16 + (lane & 15);
    const int hiHalf = lane >> 4;             // 0 | 1
    const int koffA  = hiHalf * 8;
    const int koffB  = hiHalf * 16;
    const int ncol   = tn*16 + (lane & 15);

    const _Float16* Ap = prob + (size_t)(b*QP + mrow) * PDS;
    const _Float16* A1 = k1   + (size_t)(b*QP + mrow) * PDS;
    const _Float16* A2 = k2   + (size_t)(b*QP + mrow) * PDS;
    const _Float16* A3 = k3   + (size_t)(b*QP + mrow) * PDS;
    const _Float16* B1 = t1   + (size_t)(b*NP + ncol) * PDS;
    const _Float16* B2 = t2   + (size_t)(b*NP + ncol) * PDS;
    const _Float16* B3 = t3   + (size_t)(b*NP + ncol) * PDS;

    const int kbeg = wave * (PDS / KWAVES);

    v8f accP = {}, acc1 = {}, acc2 = {}, acc3 = {};

    // ---- unroll-by-2 ping-pong pipeline; constant trip count, no guards ----
    TileSet s0 = load_tiles(Ap, A1, A2, A3, B1, B2, B3, kbeg,      koffA, koffB);
    TileSet s1 = load_tiles(Ap, A1, A2, A3, B1, B2, B3, kbeg + 32, koffA, koffB);
    int k = kbeg + 64;
    #pragma unroll 1
    for (int it = 0; it < (KSTEPS - 2) / 2; ++it) {
        wmma4(s0, accP, acc1, acc2, acc3);
        s0 = load_tiles(Ap, A1, A2, A3, B1, B2, B3, k,      koffA, koffB);
        wmma4(s1, accP, acc1, acc2, acc3);
        s1 = load_tiles(Ap, A1, A2, A3, B1, B2, B3, k + 32, koffA, koffB);
        k += 64;
    }
    wmma4(s0, accP, acc1, acc2, acc3);   // epilogue: drain both buffers
    wmma4(s1, accP, acc1, acc2, acc3);

    // ---- combine the KWAVES partial accumulators through LDS ----
    __shared__ float red[4][KWAVES][32][8];   // 16 KB
    #pragma unroll
    for (int i = 0; i < 8; ++i) {
        red[0][wave][lane][i] = accP[i];
        red[1][wave][lane][i] = acc1[i];
        red[2][wave][lane][i] = acc2[i];
        red[3][wave][lane][i] = acc3[i];
    }
    __syncthreads();
    if (wave != 0) return;

    // D layout: VGPR i = row (i + 8*hiHalf), lane&15 = col.
    const int nOut = tn*16 + (lane & 15);
    #pragma unroll
    for (int i = 0; i < 8; ++i) {
        float sP = 0.f, s1v = 0.f, s2v = 0.f, s3v = 0.f;
        #pragma unroll
        for (int w = 0; w < KWAVES; ++w) {
            sP  += red[0][w][lane][i];
            s1v += red[1][w][lane][i];
            s2v += red[2][w][lane][i];
            s3v += red[3][w][lane][i];
        }
        const int m = tm*16 + i + hiHalf*8;
        if (m < QN && nOut < NGT) {
            const float psum = qscal[(b*QP+m)*2+0];
            const float c0   = qscal[(b*QP+m)*2+1];
            const float ts   = tsum[b*NP + nOut];
            const float focal = (c0 + s1v + s2v + s3v) * (1.0f/(float)PDS);
            const float dice  = 1.0f - (2.0f*sP + 1.0f) / (psum + ts + 1.0f);
            const float* lg = logits + (((size_t)l*BN + b)*QN + m) * CC;
            const float mx = fmaxf(lg[0], lg[1]);
            const float e0 = __expf(lg[0]-mx), e1 = __expf(lg[1]-mx);
            const int cls = labels[b*NGT + nOut];
            const float pcls = (cls == 0 ? e0 : e1) / (e0 + e1);
            cost[(((size_t)l*BN + b)*QN + m)*NGT + nOut] = -pcls + focal + dice;
        }
    }
}

// ===========================================================================
// 4) Hungarian (Jonker-Volgenant, e-maxx form) on the transposed 20x100 cost.
//    One block per (l,b); column scans / argmin parallelized over 128 threads.
//    Output: matchq[l,b,n] = assigned query for gt n.
// ===========================================================================
__global__ __launch_bounds__(128)
void hungarian_kernel(const float* __restrict__ cost, int* __restrict__ matchq)
{
    const int lb  = blockIdx.x;           // 0..L*B-1
    const int tid = threadIdx.x;
    __shared__ float sC[NGT*QN];
    __shared__ float su[NGT+1];
    __shared__ float sv[QN+1];
    __shared__ float sminv[QN+1];
    __shared__ int   sp[QN+1];
    __shared__ int   sway[QN+1];
    __shared__ unsigned char sused[QN+1];
    __shared__ float rval[128];
    __shared__ int   ridx[128];
    __shared__ int   s_j0, s_j1, s_done;
    __shared__ float s_delta;

    const float* cb = cost + (size_t)lb * QN * NGT;
    for (int idx = tid; idx < NGT*QN; idx += 128) {
        const int i = idx / QN, j = idx % QN;
        sC[idx] = cb[j*NGT + i];          // transpose: rows = gt, cols = query
    }
    for (int j = tid; j <= QN; j += 128) { sv[j] = 0.f; sp[j] = 0; sway[j] = 0; }
    if (tid <= NGT) su[tid] = 0.f;
    __syncthreads();

    for (int i = 1; i <= NGT; ++i) {
        if (tid == 0) { sp[0] = i; s_j0 = 0; }
        for (int j = tid; j <= QN; j += 128) { sminv[j] = 1e30f; sused[j] = 0; }
        __syncthreads();
        while (true) {
            if (tid == 0) sused[s_j0] = 1;
            __syncthreads();
            const int j0 = s_j0;
            const int i0 = sp[j0];
            const float ui0 = su[i0];
            for (int j = tid+1; j <= QN; j += 128) {
                if (!sused[j]) {
                    const float cur = sC[(i0-1)*QN + (j-1)] - ui0 - sv[j];
                    if (cur < sminv[j]) { sminv[j] = cur; sway[j] = j0; }
                }
            }
            __syncthreads();
            float best = 1e30f; int bj = 1;
            for (int j = tid+1; j <= QN; j += 128)
                if (!sused[j] && sminv[j] < best) { best = sminv[j]; bj = j; }
            rval[tid] = best; ridx[tid] = bj;
            __syncthreads();
            for (int s = 64; s > 0; s >>= 1) {
                if (tid < s && rval[tid+s] < rval[tid]) { rval[tid] = rval[tid+s]; ridx[tid] = ridx[tid+s]; }
                __syncthreads();
            }
            if (tid == 0) { s_j1 = ridx[0]; s_delta = rval[0]; }
            __syncthreads();
            const float delta = s_delta;
            for (int j = tid; j <= QN; j += 128) {
                if (sused[j]) { su[sp[j]] += delta; sv[j] -= delta; }
                else          { sminv[j] -= delta; }
            }
            __syncthreads();
            if (tid == 0) { s_j0 = s_j1; s_done = (sp[s_j1] == 0); }
            __syncthreads();
            if (s_done) break;
        }
        if (tid == 0) {
            int j0 = s_j0;
            while (j0) { const int j1 = sway[j0]; sp[j0] = sp[j1]; j0 = j1; }
        }
        __syncthreads();
    }
    for (int j = tid+1; j <= QN; j += 128)
        if (sp[j] > 0) matchq[lb*NGT + (sp[j]-1)] = j - 1;
}

// ===========================================================================
// 5) Matched-pair full-res loss: fused bilinear 128->512 upsample of the
//    matched query mask; accumulate focal sum, inter, prob sum, gt sum.
//    One 256-thread block per (l,b,n); prefetch the streaming gt rows.
// ===========================================================================
__global__ __launch_bounds__(256)
void pair_loss_kernel(const float* __restrict__ pm, const int* __restrict__ gtm,
                      const int* __restrict__ matchq, float* __restrict__ pair)
{
    const int idx = blockIdx.x;                 // l*B*N + b*N + n
    const int l = idx / (BN*NGT);
    const int r = idx % (BN*NGT);
    const int b = r / NGT, n = r % NGT;
    const int q = matchq[(l*BN + b)*NGT + n];
    const float* src = pm + (((size_t)l*BN + b)*QN + q) * (HSML*HSML);
    const int*   g   = gtm + ((size_t)b*NGT + n) * (HREZ*HREZ);

    float fs = 0.f, inter = 0.f, ps = 0.f, ts = 0.f;
    for (int p = threadIdx.x; p < HREZ*HREZ; p += blockDim.x) {
        __builtin_prefetch(g + p + 4*256, 0, 0);    // stream-ahead on gt
        const int y = p >> 9, x = p & 511;
        const float sy = (y + 0.5f)*0.25f - 0.5f;
        const float sx = (x + 0.5f)*0.25f - 0.5f;
        const int y0 = (int)floorf(sy); const float wy = sy - (float)y0;
        const int x0 = (int)floorf(sx); const float wx = sx - (float)x0;
        const int y0c = y0   < 0 ? 0 : (y0   > HSML-1 ? HSML-1 : y0);
        const int y1c = y0+1 < 0 ? 0 : (y0+1 > HSML-1 ? HSML-1 : y0+1);
        const int x0c = x0   < 0 ? 0 : (x0   > HSML-1 ? HSML-1 : x0);
        const int x1c = x0+1 < 0 ? 0 : (x0+1 > HSML-1 ? HSML-1 : x0+1);
        const float v = (1.f-wy)*((1.f-wx)*src[y0c*HSML+x0c] + wx*src[y0c*HSML+x1c])
                      +       wy*((1.f-wx)*src[y1c*HSML+x0c] + wx*src[y1c*HSML+x1c]);
        const float t  = (float)g[p];
        const float pr = 1.f / (1.f + __expf(-v));
        const float bce = fmaxf(v, 0.f) - v*t + log1pf(__expf(-fabsf(v)));
        const float p_t = pr*t + (1.f-pr)*(1.f-t);
        const float at  = ALPHA_F*t + (1.f-ALPHA_F)*(1.f-t);
        const float omp = 1.f - p_t;
        fs    += at*omp*omp*bce;
        inter += pr*t; ps += pr; ts += t;
    }
    __shared__ float sr[256*4];
    const int tid = threadIdx.x;
    sr[tid*4+0]=fs; sr[tid*4+1]=inter; sr[tid*4+2]=ps; sr[tid*4+3]=ts;
    __syncthreads();
    for (int s = 128; s > 0; s >>= 1) {
        if (tid < s)
            #pragma unroll
            for (int c = 0; c < 4; ++c) sr[tid*4+c] += sr[(tid+s)*4+c];
        __syncthreads();
    }
    if (tid == 0) {
        float* o = pair + (size_t)idx*4;
        o[0]=sr[0]; o[1]=sr[1]; o[2]=sr[2]; o[3]=sr[3];
    }
}

// ===========================================================================
// 6) Finalize: weighted CE over all (l,b,q) + mean focal/dice over pairs.
// ===========================================================================
__global__ __launch_bounds__(256)
void finalize_kernel(const float* __restrict__ logits, const int* __restrict__ labels,
                     const int* __restrict__ matchq, const float* __restrict__ pair,
                     float* __restrict__ out)
{
    __shared__ float s_wnll[LN], s_wsum[LN], s_f[LN], s_d[LN];
    const int tid = threadIdx.x;
    if (tid < LN) { s_wnll[tid]=0.f; s_wsum[tid]=0.f; s_f[tid]=0.f; s_d[tid]=0.f; }
    __syncthreads();

    for (int i = tid; i < LN*BN*QN; i += blockDim.x) {
        const int l = i / (BN*QN);
        const int r = i % (BN*QN);
        const int b = r / QN, q = r % QN;
        int tgt = CC - 1;                              // "no object"
        for (int n = 0; n < NGT; ++n)
            if (matchq[(l*BN+b)*NGT + n] == q) tgt = labels[b*NGT + n];
        const float* lg = logits + (size_t)i * CC;
        const float mx = fmaxf(lg[0], lg[1]);
        const float lse = mx + __logf(__expf(lg[0]-mx) + __expf(lg[1]-mx));
        const float nll = lse - lg[tgt];
        const float w = (tgt == CC-1) ? 0.1f : 1.0f;
        atomicAdd(&s_wnll[l], w*nll);
        atomicAdd(&s_wsum[l], w);
    }
    for (int i = tid; i < LN*BN*NGT; i += blockDim.x) {
        const int l = i / (BN*NGT);
        const float* pp = pair + (size_t)i*4;
        atomicAdd(&s_f[l], pp[0]);
        const float dice = 1.f - (2.f*pp[1] + 1.f) / (pp[2] + pp[3] + 1.f);
        atomicAdd(&s_d[l], dice);
    }
    __syncthreads();
    if (tid == 0) {
        float total = 0.f;
        for (int l = 0; l < LN; ++l) {
            const float ce    = s_wnll[l] / s_wsum[l];
            const float focal = s_f[l] / (float)(BN*NGT) / (float)(HREZ*HREZ);
            const float dice  = s_d[l] / (float)(BN*NGT);
            total += 2.0f*ce + 5.0f*focal + 5.0f*dice;
        }
        out[0] = total;
    }
}

// ===========================================================================
extern "C" void kernel_launch(void* const* d_in, const int* in_sizes, int n_in,
                              void* d_out, int out_size, void* d_ws, size_t ws_size,
                              hipStream_t stream)
{
    (void)in_sizes; (void)n_in; (void)out_size; (void)ws_size;
    const float* pred_logits = (const float*)d_in[0];   // [L,B,Q,C]
    const float* pred_masks  = (const float*)d_in[1];   // [L,B,Q,128,128]
    const int*   gt_labels   = (const int*)d_in[2];     // [B,N]
    const int*   gt_masks    = (const int*)d_in[3];     // [B,N,512,512]
    float* out = (float*)d_out;

    char* ws = (char*)d_ws;
    _Float16* t1   = (_Float16*)(ws + OFF_T1);
    _Float16* t2   = (_Float16*)(ws + OFF_T2);
    _Float16* t3   = (_Float16*)(ws + OFF_T3);
    float*    tsum = (float*)   (ws + OFF_TSUM);
    _Float16* prob = (_Float16*)(ws + OFF_PROB);
    _Float16* k1   = (_Float16*)(ws + OFF_K1);
    _Float16* k2   = (_Float16*)(ws + OFF_K2);
    _Float16* k3   = (_Float16*)(ws + OFF_K3);
    float*    qsc  = (float*)   (ws + OFF_QSC);
    float*    cost = (float*)   (ws + OFF_COST);
    int*      mq   = (int*)     (ws + OFF_MATCH);
    float*    pair = (float*)   (ws + OFF_PAIR);

    // 1) gt downsample + t powers (shared by all layers)
    prep_gt_kernel<<<BN*NP, 256, 0, stream>>>(gt_masks, t1, t2, t3, tsum);

    // 2) per-layer: pred prep then WMMA cost matrix (prob/k buffers reused;
    //    stream ordering serializes correctly)
    for (int l = 0; l < LN; ++l) {
        prep_pred_kernel<<<BN*QP, 256, 0, stream>>>(l, pred_masks, prob, k1, k2, k3, qsc);
        cost_wmma_kernel<<<dim3(QP/16, NP/16, BN), 32*KWAVES, 0, stream>>>(
            l, prob, k1, k2, k3, t1, t2, t3, qsc, tsum,
            pred_logits, gt_labels, cost);
    }

    // 3) Hungarian matching, all (l,b) problems in parallel
    hungarian_kernel<<<LN*BN, 128, 0, stream>>>(cost, mq);

    // 4) full-res matched-pair reductions
    pair_loss_kernel<<<LN*BN*NGT, 256, 0, stream>>>(pred_masks, gt_masks, mq, pair);

    // 5) combine everything into the scalar loss
    finalize_kernel<<<1, 256, 0, stream>>>(pred_logits, gt_labels, mq, pair, out);
}
